// RadialAttention_35321811042709
// MI455X (gfx1250) — matrix operand order repaired and checked by reading, not compiled, not run on hardware
//
#include <hip/hip_runtime.h>
#include <math.h>

#define SEQ 4096
#define NH  8
#define HD  32
#define EMB 256

typedef _Float16 f16;
typedef __attribute__((ext_vector_type(16))) _Float16 v16h;
typedef __attribute__((ext_vector_type(8)))  _Float16 v8h;
typedef __attribute__((ext_vector_type(8)))  float    v8f;

static __device__ __forceinline__ v16h join16(v8h lo, v8h hi) {
  return __builtin_shufflevector(lo, hi, 0,1,2,3,4,5,6,7,8,9,10,11,12,13,14,15);
}

static __device__ __forceinline__ v8f wmma_f16(v16h a, v16h b, v8f c) {
  // D = A(16x32 f16) x B(32x16 f16) + C(16x16 f32)
  return __builtin_amdgcn_wmma_f32_16x16x32_f16(false, a, false, b, (short)0, c,
                                                false, false);
}

static __device__ __forceinline__ float fast_exp2(float x) {
#if __has_builtin(__builtin_amdgcn_exp2f)
  return __builtin_amdgcn_exp2f(x);
#else
  return exp2f(x);
#endif
}

// 16-byte global -> LDS copy: async (ASYNCcnt-tracked) when the gfx1250
// builtin is available, synchronous vector copy otherwise.
// Probe-confirmed param 0 type: 'int __attribute__((vector_size(16))) __device__ *'
// i.e. non-const v4i in address space 1; LDS destination in address space 3.
static __device__ __forceinline__ void stage_copy16(const f16* g, f16* l) {
#if __has_builtin(__builtin_amdgcn_global_load_async_to_lds_b128)
  typedef int v4i __attribute__((vector_size(16)));
  typedef __attribute__((address_space(1))) v4i* gp_t;
  typedef __attribute__((address_space(3))) v4i* lp_t;
  gp_t gp = (gp_t)(v4i*)(void*)const_cast<f16*>(g);
  lp_t lp = (lp_t)(v4i*)(void*)l;
  __builtin_amdgcn_global_load_async_to_lds_b128(gp, lp, 0, 0);
#else
  *(v8h*)l = *(const v8h*)g;
#endif
}

static __device__ __forceinline__ void wait_async_prev() {
#if __has_builtin(__builtin_amdgcn_global_load_async_to_lds_b128)
# if __has_builtin(__builtin_amdgcn_s_wait_asynccnt)
  __builtin_amdgcn_s_wait_asynccnt(4);   // this wave's previous stage (4 ops) done
# else
  asm volatile("s_wait_asynccnt 0x4" ::: "memory");
# endif
#endif
}

static __device__ __forceinline__ void wait_async_all() {
#if __has_builtin(__builtin_amdgcn_global_load_async_to_lds_b128)
# if __has_builtin(__builtin_amdgcn_s_wait_asynccnt)
  __builtin_amdgcn_s_wait_asynccnt(0);
# else
  asm volatile("s_wait_asynccnt 0x0" ::: "memory");
# endif
#endif
}

// ---------------------------------------------------------------------------
// Kernel 1: fused QKV projection (f32 -> f16 WMMA, f32 accumulate).
// Q,K stored head-major [H][S][32]; V stored TRANSPOSED [H][32][S] with the
// radial mask folded in, so the flash kernel's V^T A-fragments are contiguous.
// ---------------------------------------------------------------------------
__global__ __launch_bounds__(128)
void qkv_proj_kernel(const float* __restrict__ query,
                     const float* __restrict__ key_,
                     const float* __restrict__ value,
                     const float* __restrict__ radial,
                     const float* __restrict__ Wq,
                     const float* __restrict__ Wk,
                     const float* __restrict__ Wv,
                     f16* __restrict__ Qh, f16* __restrict__ Kh,
                     f16* __restrict__ Vt)
{
  const int wave = threadIdx.x >> 5;
  const int lane = threadIdx.x & 31;
  int t = blockIdx.x * 4 + wave;
  const int tiles_per_mat = (SEQ / 16) * (EMB / 16);
  const int mat = t / tiles_per_mat;           // 0=Q 1=K 2=V
  t %= tiles_per_mat;
  const int mt = t / (EMB / 16);
  const int nt = t % (EMB / 16);
  const int m0 = mt * 16, n0 = nt * 16;

  const float* A = (mat == 0) ? query : (mat == 1) ? key_ : value;
  const float* W = (mat == 0) ? Wq    : (mat == 1) ? Wk   : Wv;

  const int col  = lane & 15;
  const int hilo = lane >> 4;                  // lane half
  const float* Arow = A + (size_t)(m0 + col) * EMB;
  const float* Wrow = W + (size_t)(n0 + col) * EMB;
  const int kha = hilo * 8;                    // unused (A-frag also uses khb)
  const int khb = hilo * 16;                   // B-frag K sub-offset

  v8f acc = {};
  #pragma unroll
  for (int k0 = 0; k0 < EMB; k0 += 32) {
    v16h a, b;
    #pragma unroll
    for (int i = 0; i < 8; ++i) {
      a[i]     = (f16)Arow[k0 + khb + i];
      a[8 + i] = (f16)Arow[k0 + khb + 8 + i];
    }
    #pragma unroll
    for (int i = 0; i < 16; ++i)
      b[i] = (f16)Wrow[k0 + khb + i];
    acc = wmma_f16(a, b, acc);
  }

  const int head  = nt >> 1;
  const int od    = (nt & 1) * 16 + col;       // head-dim index 0..31
  const int rbase = hilo * 8;
  if (mat == 2) {
    // transposed store + radial mask: Vt[head][od][s], contiguous in s
    f16* dst = Vt + ((size_t)head * HD + od) * SEQ + (m0 + rbase);
    #pragma unroll
    for (int j = 0; j < 8; ++j)
      dst[j] = (f16)(acc[j] * radial[m0 + rbase + j]);
  } else {
    f16* Dst = (mat == 0) ? Qh : Kh;
    #pragma unroll
    for (int j = 0; j < 8; ++j)
      Dst[((size_t)head * SEQ + (m0 + rbase + j)) * HD + od] = (f16)acc[j];
  }
}

// ---------------------------------------------------------------------------
// Kernel 2: flash attention with double-buffered async LDS staging.
// Each wave owns 16 query rows of one head; the block shares K/V^T KV-blocks
// of 64 staged in LDS. S^T = K x Q^T (softmax rows live in lane columns,
// reduced with one shfl_xor(16)); O^T = V^T x P^T accumulated in f32.
// Softmax runs in base-2 (v_exp_f32 native) with log2(e) folded into scale.
// ---------------------------------------------------------------------------
__global__ __launch_bounds__(128)
void flash_attn_kernel(const f16* __restrict__ Qh,
                       const f16* __restrict__ Kh,
                       const f16* __restrict__ Vt,
                       f16* __restrict__ attn)
{
  __shared__ __attribute__((aligned(16))) f16 kbuf[2][64 * HD];  // [64 kv][32 d]
  __shared__ __attribute__((aligned(16))) f16 vbuf[2][HD * 64];  // [32 d][64 kv]

  const int tid  = threadIdx.x;
  const int wave = tid >> 5;
  const int lane = tid & 31;
  const int col  = lane & 15;
  const int hilo = lane >> 4;
  const int h    = blockIdx.y;
  const int q0   = blockIdx.x * 64 + wave * 16;

  const f16* Kp = Kh + (size_t)h * SEQ * HD;
  const f16* Vp = Vt + (size_t)h * HD * SEQ;

  // Q as B-operand: lane = query column, d = 8h+s and 16+8h+(s-8)
  const f16* qrow = Qh + ((size_t)h * SEQ + (q0 + col)) * HD + hilo * 8;
  const v16h qb = join16(*(const v8h*)qrow, *(const v8h*)(qrow + 16));

  v8f o0 = {}, o1 = {};
  float m = -3.0e38f, l = 0.f;
  const float sc2 = 0.2550347282f;             // log2(e)/sqrt(32)

  const int c0 = tid, c1 = tid + 128;          // 16B chunk ids (256 per buffer)

  // prologue: stage KV block 0 into buffer 0
  stage_copy16(Kp + c0 * 8, &kbuf[0][c0 * 8]);
  stage_copy16(Kp + c1 * 8, &kbuf[0][c1 * 8]);
  stage_copy16(Vp + (size_t)(c0 >> 3) * SEQ + (c0 & 7) * 8, &vbuf[0][c0 * 8]);
  stage_copy16(Vp + (size_t)(c1 >> 3) * SEQ + (c1 & 7) * 8, &vbuf[0][c1 * 8]);

  const int nIter = SEQ / 64;
  for (int it = 0; it < nIter; ++it) {
    const int b = it & 1;
    if (it + 1 < nIter) {
      const int j0n = (it + 1) * 64;
      stage_copy16(Kp + (size_t)j0n * HD + c0 * 8, &kbuf[b ^ 1][c0 * 8]);
      stage_copy16(Kp + (size_t)j0n * HD + c1 * 8, &kbuf[b ^ 1][c1 * 8]);
      stage_copy16(Vp + (size_t)(c0 >> 3) * SEQ + j0n + (c0 & 7) * 8,
                   &vbuf[b ^ 1][c0 * 8]);
      stage_copy16(Vp + (size_t)(c1 >> 3) * SEQ + j0n + (c1 & 7) * 8,
                   &vbuf[b ^ 1][c1 * 8]);
      wait_async_prev();                       // current buffer landed
    } else {
      wait_async_all();
    }
    __syncthreads();                           // all waves' stages visible

    #pragma unroll
    for (int sub = 0; sub < 64; sub += 32) {
      // two K row-tiles as A-operands, from LDS
      const f16* kr0 = &kbuf[b][(sub + col) * HD + hilo * 8];
      const f16* kr1 = kr0 + 16 * HD;
      const v16h ka0 = join16(*(const v8h*)kr0, *(const v8h*)(kr0 + 16));
      const v16h ka1 = join16(*(const v8h*)kr1, *(const v8h*)(kr1 + 16));

      const v8f z = {};
      v8f s0 = wmma_f16(ka0, qb, z);           // S^T tile, kv sub..sub+15
      v8f s1 = wmma_f16(ka1, qb, z);           // S^T tile, kv sub+16..sub+31

      // online softmax in base-2 over kv (8 in-lane rows + partner half)
      float tm = -3.0e38f;
      #pragma unroll
      for (int j = 0; j < 8; ++j) {
        s0[j] *= sc2; s1[j] *= sc2;
        tm = fmaxf(tm, fmaxf(s0[j], s1[j]));
      }
      tm = fmaxf(tm, __shfl_xor(tm, 16, 32));
      const float mn    = fmaxf(m, tm);
      const float alpha = fast_exp2(m - mn);

      float ps = 0.f;
      unsigned int t0[4], t1[4];
      #pragma unroll
      for (int j = 0; j < 4; ++j) {
        const float e00 = fast_exp2(s0[2*j]     - mn);
        const float e01 = fast_exp2(s0[2*j + 1] - mn);
        const float e10 = fast_exp2(s1[2*j]     - mn);
        const float e11 = fast_exp2(s1[2*j + 1] - mn);
        ps += (e00 + e01) + (e10 + e11);
        union { f16 hh[2]; unsigned int u; } p;
        p.hh[0] = (f16)e00; p.hh[1] = (f16)e01; t0[j] = p.u;
        p.hh[0] = (f16)e10; p.hh[1] = (f16)e11; t1[j] = p.u;
      }
      ps += __shfl_xor(ps, 16, 32);
      l = l * alpha + ps;
      m = mn;

      // repack P^T (D layout) into B-operand layout via lane-half exchange
      unsigned int pb[8];
      #pragma unroll
      for (int j = 0; j < 4; ++j) {
        const unsigned int x0 = __shfl_xor(t0[j], 16, 32);
        const unsigned int x1 = __shfl_xor(t1[j], 16, 32);
        pb[j]     = hilo ? x1    : t0[j];
        pb[4 + j] = hilo ? t1[j] : x0;
      }
      v16h pbf;
      #pragma unroll
      for (int j = 0; j < 8; ++j) {
        union { unsigned int u; f16 hh[2]; } p; p.u = pb[j];
        pbf[2*j] = p.hh[0]; pbf[2*j + 1] = p.hh[1];
      }

      // V^T A-fragments from LDS (contiguous thanks to transposed store)
      const f16* vr0 = &vbuf[b][col * 64 + sub + hilo * 16];
      const f16* vr1 = vr0 + 16 * 64;
      const v16h va0 = join16(*(const v8h*)vr0, *(const v8h*)(vr0 + 8));
      const v16h va1 = join16(*(const v8h*)vr1, *(const v8h*)(vr1 + 8));

      #pragma unroll
      for (int j = 0; j < 8; ++j) { o0[j] *= alpha; o1[j] *= alpha; }
      o0 = wmma_f16(va0, pbf, o0);             // O^T rows d 0..15
      o1 = wmma_f16(va1, pbf, o1);             // O^T rows d 16..31
    }
    __syncthreads();                           // reads done before overwrite
  }

  const float linv = 1.0f / l;
  // O^T lane column = query row; VGPR index = head-dim -> contiguous stores
  f16* dst = attn + (size_t)(q0 + col) * (NH * HD) + h * HD + hilo * 8;
  #pragma unroll
  for (int j = 0; j < 8; ++j) {
    dst[j]      = (f16)(o0[j] * linv);
    dst[16 + j] = (f16)(o1[j] * linv);
  }
}

// ---------------------------------------------------------------------------
// Kernel 3: output projection attn(f16) @ Wo^T -> f32 d_out.
// ---------------------------------------------------------------------------
__global__ __launch_bounds__(128)
void out_proj_kernel(const f16* __restrict__ attn,
                     const float* __restrict__ Wo,
                     float* __restrict__ out)
{
  const int wave = threadIdx.x >> 5;
  const int lane = threadIdx.x & 31;
  int t = blockIdx.x * 4 + wave;
  const int mt = t / (EMB / 16);
  const int nt = t % (EMB / 16);
  const int m0 = mt * 16, n0 = nt * 16;
  const int col = lane & 15, hilo = lane >> 4;

  const f16*   Arow = attn + (size_t)(m0 + col) * EMB + hilo * 16;
  const float* Wrow = Wo   + (size_t)(n0 + col) * EMB + hilo * 16;

  v8f acc = {};
  #pragma unroll
  for (int k0 = 0; k0 < EMB; k0 += 32) {
    const v16h a = join16(*(const v8h*)(Arow + k0),
                          *(const v8h*)(Arow + k0 + 8));
    v16h b;
    #pragma unroll
    for (int i = 0; i < 16; ++i) b[i] = (f16)Wrow[k0 + i];
    acc = wmma_f16(a, b, acc);
  }
  float* dst = out + (size_t)(m0 + hilo * 8) * EMB + n0 + col;
  #pragma unroll
  for (int j = 0; j < 8; ++j) dst[j * EMB] = acc[j];
}

// ---------------------------------------------------------------------------
extern "C" void kernel_launch(void* const* d_in, const int* in_sizes, int n_in,
                              void* d_out, int out_size, void* d_ws, size_t ws_size,
                              hipStream_t stream) {
  const float* query  = (const float*)d_in[0];
  const float* key_   = (const float*)d_in[1];
  const float* value  = (const float*)d_in[2];
  const float* radial = (const float*)d_in[3];
  const float* Wq     = (const float*)d_in[4];
  const float* Wk     = (const float*)d_in[5];
  const float* Wv     = (const float*)d_in[6];
  const float* Wo     = (const float*)d_in[7];
  float* out = (float*)d_out;

  char* ws = (char*)d_ws;
  f16* Qh   = (f16*)(ws);                                  // 2 MB [H][S][32]
  f16* Kh   = (f16*)(ws + (size_t)2 * 1024 * 1024);        // 2 MB [H][S][32]
  f16* Vt   = (f16*)(ws + (size_t)4 * 1024 * 1024);        // 2 MB [H][32][S]
  f16* attn = (f16*)(ws + (size_t)6 * 1024 * 1024);        // 2 MB [S][256]

  qkv_proj_kernel<<<3 * (SEQ / 16) * (EMB / 16) / 4, 128, 0, stream>>>(
      query, key_, value, radial, Wq, Wk, Wv, Qh, Kh, Vt);
  flash_attn_kernel<<<dim3(SEQ / 64, NH), 128, 0, stream>>>(Qh, Kh, Vt, attn);
  out_proj_kernel<<<(SEQ / 16) * (EMB / 16) / 4, 128, 0, stream>>>(attn, Wo, out);
}
